// ImprovedFragranceGNN_46755013984593
// MI455X (gfx1250) — matrix operand order repaired and checked by reading, not compiled, not run on hardware
//
#include <hip/hip_runtime.h>
#include <hip/hip_bf16.h>
#include <cstdint>

// ---------------- problem constants (from reference) ----------------
#define NN   200000
#define EE   3200000
#define GG   8192
#define IND  128
#define HID  256
#define H2   128
#define NL   128

// ---------------- types ----------------
typedef __attribute__((ext_vector_type(16))) __bf16 v16bf;
typedef __attribute__((ext_vector_type(8)))  __bf16 v8bf;
typedef __attribute__((ext_vector_type(4)))  __bf16 v4bf;
typedef __attribute__((ext_vector_type(8)))  float  v8f;

__device__ __forceinline__ __bf16 f2bf(float f) {
    union { float f; unsigned u; } a; a.f = f;
    unsigned u = a.u;
    if ((u & 0x7fffffffu) > 0x7f800000u) u = 0x7fc00000u;      // NaN
    else u = u + 0x7fffu + ((u >> 16) & 1u);                   // RNE
    union { unsigned short s; __bf16 b; } c; c.s = (unsigned short)(u >> 16);
    return c.b;
}

// gfx1250 async copy global -> LDS (ASYNCcnt tracked)
__device__ __forceinline__ void async_b128(void* lds, const void* g) {
    unsigned l = (unsigned)(uintptr_t)lds;   // LDS aperture: addr[31:0] == wave LDS offset
    asm volatile("global_load_async_to_lds_b128 %0, %1, off" :: "v"(l), "v"(g) : "memory");
}
__device__ __forceinline__ void wait_async0() {
    asm volatile("s_wait_asynccnt 0x0" ::: "memory");
}

// ---------------- tiny elementwise kernels ----------------
__global__ void k_fill(float* __restrict__ p, float v, int n) {
    int i = blockIdx.x * blockDim.x + threadIdx.x;
    if (i < n) p[i] = v;
}

__global__ void k_deg(const int* __restrict__ col, float* __restrict__ deg, int e) {
    int i = blockIdx.x * blockDim.x + threadIdx.x;
    if (i < e) atomicAdd(&deg[col[i]], 1.0f);
}

__global__ void k_rsqrt(float* __restrict__ p, int n) {
    int i = blockIdx.x * blockDim.x + threadIdx.x;
    if (i < n) { float v = p[i]; p[i] = (v > 0.f) ? rsqrtf(v) : 0.f; }
}

// f32 -> bf16 (4 per thread)
__global__ void k_cvt(const float* __restrict__ x, __bf16* __restrict__ y, long n4) {
    long t = (long)blockIdx.x * blockDim.x + threadIdx.x;
    if (t >= n4) return;
    long i = t * 4;
    float4 v = *(const float4*)(x + i);
    v4bf o; o[0] = f2bf(v.x); o[1] = f2bf(v.y); o[2] = f2bf(v.z); o[3] = f2bf(v.w);
    *(v4bf*)(y + i) = o;
}

// f32 -> bf16 with per-column affine (fused batchnorm apply)
__global__ void k_cvt_affine(const float* __restrict__ x, const float* __restrict__ scale,
                             const float* __restrict__ shift, __bf16* __restrict__ y,
                             long n4, int colmask) {
    long t = (long)blockIdx.x * blockDim.x + threadIdx.x;
    if (t >= n4) return;
    long i = t * 4;
    int c = (int)(i & colmask);
    float4 v = *(const float4*)(x + i);
    v4bf o;
    o[0] = f2bf(v.x * scale[c+0] + shift[c+0]);
    o[1] = f2bf(v.y * scale[c+1] + shift[c+1]);
    o[2] = f2bf(v.z * scale[c+2] + shift[c+2]);
    o[3] = f2bf(v.w * scale[c+3] + shift[c+3]);
    *(v4bf*)(y + i) = o;
}

// W[K,Nc] f32 -> Wt[Nc,K] bf16
__global__ void k_wt(const float* __restrict__ W, __bf16* __restrict__ Wt, int K, int Nc) {
    int t = blockIdx.x * blockDim.x + threadIdx.x;
    if (t >= K * Nc) return;
    int n = t / K, k = t - n * K;
    Wt[t] = f2bf(W[(long)k * Nc + n]);
}

// ---------------- WMMA bf16 GEMM: C[M,*] = A[M,K] @ Wt[Nc,K]^T (+bias)(+relu) ----
// block tile 128x128, 8 waves (4 row groups x 2 col groups), each wave 2x4 tiles
// of 16x16, K-step 32, double-buffered LDS filled with async global->LDS copies.
#define BM 128
#define BN 128
#define BK 32
#define LDA 40
#define LDB 40

__global__ __launch_bounds__(256) void k_gemm_bf16(
    const __bf16* __restrict__ A, const __bf16* __restrict__ Bt,
    const float* __restrict__ bias, float* __restrict__ C,
    int M, int K, int ldc, int relu)
{
    __shared__ __align__(16) __bf16 sA[2][BM * LDA];
    __shared__ __align__(16) __bf16 sB[2][BN * LDB];

    const int tid  = threadIdx.x;
    const int lane = tid & 31;
    const int wid  = tid >> 5;
    const int wr   = wid & 3;       // wave row group: rows wr*32
    const int wc   = wid >> 2;      // wave col group: cols wc*64
    const int m0   = blockIdx.x * BM;
    const int n0   = blockIdx.y * BN;
    const int half = lane >> 4;     // 0/1
    const int lrow = lane & 15;

    // per-thread staging coordinates: 2 chunks of 8 bf16 for A, 2 for B
    const int r0  = tid >> 2;             // chunk tid      -> row tid/4
    const int kc0 = (tid & 3) * 8;
    const int r1  = (tid + 256) >> 2;     // chunk tid+256
    const int kc1 = ((tid + 256) & 3) * 8;

    auto stage = [&](int k0, int buf) {
        int ga0 = m0 + r0; if (ga0 >= M) ga0 = M - 1;   // clamp (results discarded)
        int ga1 = m0 + r1; if (ga1 >= M) ga1 = M - 1;
        async_b128(&sA[buf][r0 * LDA + kc0], A  + (long)ga0 * K + k0 + kc0);
        async_b128(&sA[buf][r1 * LDA + kc1], A  + (long)ga1 * K + k0 + kc1);
        async_b128(&sB[buf][r0 * LDB + kc0], Bt + (long)(n0 + r0) * K + k0 + kc0);
        async_b128(&sB[buf][r1 * LDB + kc1], Bt + (long)(n0 + r1) * K + k0 + kc1);
    };

    v8f acc[2][4] = {};
    const int iters = K / BK;

    stage(0, 0);
    for (int it = 0; it < iters; ++it) {
        const int buf = it & 1;
        wait_async0();        // our async fills of `buf` are complete
        __syncthreads();      // everyone's fills complete; prior reads of `buf` done
        if (it + 1 < iters) stage((it + 1) * BK, buf ^ 1);  // overlap next copy

        // A fragment: lane holds row (lrow); K in {kh..kh+7} U {kh+16..kh+23}
        const int kh = half * 8;
        v16bf af[2], bfr[4];
        #pragma unroll
        for (int mt = 0; mt < 2; ++mt) {
            int r = wr * 32 + mt * 16 + lrow;
            const v8bf p0 = *(const v8bf*)&sA[buf][r * LDA + kh];
            const v8bf p1 = *(const v8bf*)&sA[buf][r * LDA + kh + 16];
            #pragma unroll
            for (int i = 0; i < 8; ++i) { af[mt][i] = p0[i]; af[mt][i + 8] = p1[i]; }
        }
        // B fragment: lane holds col (lrow); K in {kb..kb+7} U {kb+16..kb+23}
        const int kb = half * 8;
        #pragma unroll
        for (int nt = 0; nt < 4; ++nt) {
            int r = wc * 64 + nt * 16 + lrow;
            const v8bf q0 = *(const v8bf*)&sB[buf][r * LDB + kb];
            const v8bf q1 = *(const v8bf*)&sB[buf][r * LDB + kb + 16];
            #pragma unroll
            for (int i = 0; i < 8; ++i) { bfr[nt][i] = q0[i]; bfr[nt][i + 8] = q1[i]; }
        }

        #pragma unroll
        for (int mt = 0; mt < 2; ++mt)
            #pragma unroll
            for (int nt = 0; nt < 4; ++nt)
                acc[mt][nt] = __builtin_amdgcn_wmma_f32_16x16x32_bf16(
                    false, af[mt], false, bfr[nt], (short)0, acc[mt][nt], false, false);
    }

    // store: D element (m,n): lane = 16*(m>=8)+n within tile, vgpr v = m&7
    #pragma unroll
    for (int mt = 0; mt < 2; ++mt) {
        #pragma unroll
        for (int nt = 0; nt < 4; ++nt) {
            int gn = n0 + wc * 64 + nt * 16 + lrow;
            float b = bias ? bias[gn] : 0.0f;
            #pragma unroll
            for (int v = 0; v < 8; ++v) {
                int gm = m0 + wr * 32 + mt * 16 + half * 8 + v;
                if (gm < M) {
                    float val = acc[mt][nt][v] + b;
                    if (relu) val = fmaxf(val, 0.0f);
                    C[(long)gm * ldc + gn] = val;
                }
            }
        }
    }
}

// ---------------- edge scatter: agg[col] += h[row] * dinv[row]*dinv[col] -------
__global__ void k_scatter(const float* __restrict__ h, const int* __restrict__ row,
                          const int* __restrict__ col, const float* __restrict__ dinv,
                          float* __restrict__ agg, int e, int cptl2, int cptmask, int C)
{
    long t = (long)blockIdx.x * blockDim.x + threadIdx.x;
    long ei = t >> cptl2;
    int  c  = (int)(t & cptmask) << 2;
    if (ei >= e) return;
    int r = row[ei], cl = col[ei];
    if (ei + 8 < e) __builtin_prefetch(h + (long)row[ei + 8] * C + c, 0, 1);
    float nrm = dinv[r] * dinv[cl];
    float4 hv = *(const float4*)(h + (long)r * C + c);
    float* dst = agg + (long)cl * C + c;
    atomicAdd(dst + 0, hv.x * nrm);
    atomicAdd(dst + 1, hv.y * nrm);
    atomicAdd(dst + 2, hv.z * nrm);
    atomicAdd(dst + 3, hv.w * nrm);
}

// h = relu(agg + h*dinv^2 + bias)   (in place over h)
__global__ void k_selfbias(float* __restrict__ h, const float* __restrict__ agg,
                           const float* __restrict__ dinv, const float* __restrict__ bias,
                           long n4, int cptl2, int cptmask)
{
    long t = (long)blockIdx.x * blockDim.x + threadIdx.x;
    if (t >= n4) return;
    long i = t >> cptl2;
    int  c = (int)(t & cptmask) << 2;
    long idx = t * 4;
    float d = dinv[i]; float s = d * d;
    float4 hv = *(const float4*)(h + idx);
    float4 av = *(const float4*)(agg + idx);
    float4 bv = *(const float4*)(bias + c);
    float4 o;
    o.x = fmaxf(av.x + hv.x * s + bv.x, 0.f);
    o.y = fmaxf(av.y + hv.y * s + bv.y, 0.f);
    o.z = fmaxf(av.z + hv.z * s + bv.z, 0.f);
    o.w = fmaxf(av.w + hv.w * s + bv.w, 0.f);
    *(float4*)(h + idx) = o;
}

// ---------------- mean pool (C=128, into concat slice with stride 512) ---------
__global__ void k_pool_sum(const float* __restrict__ h, const int* __restrict__ batch,
                           float* __restrict__ slice, float* __restrict__ cnt, int n)
{
    long t = (long)blockIdx.x * blockDim.x + threadIdx.x;
    long i = t >> 5;
    int  c = (int)(t & 31) << 2;
    if (i >= n) return;
    int g = batch[i];
    float4 hv = *(const float4*)(h + i * 128 + c);
    float* dst = slice + (long)g * 512 + c;
    atomicAdd(dst + 0, hv.x);
    atomicAdd(dst + 1, hv.y);
    atomicAdd(dst + 2, hv.z);
    atomicAdd(dst + 3, hv.w);
    if (c == 0) atomicAdd(&cnt[g], 1.0f);
}

__global__ void k_pool_div(float* __restrict__ slice, const float* __restrict__ cnt) {
    int t = blockIdx.x * blockDim.x + threadIdx.x;   // G*128 threads
    int g = t >> 7, c = t & 127;
    if (g < GG) slice[(long)g * 512 + c] /= fmaxf(cnt[g], 1.0f);
}

// ---------------- batchnorm stats -> per-column scale/shift --------------------
__global__ void k_bn_stats(const float* __restrict__ x, const float* __restrict__ gamma,
                           const float* __restrict__ beta, float* __restrict__ scale,
                           float* __restrict__ shift, int rows, int cols)
{
    int c = blockIdx.x;
    __shared__ float ssum[256], ssq[256];
    float s = 0.f, q = 0.f;
    for (int r = threadIdx.x; r < rows; r += 256) {
        float v = x[(long)r * cols + c];
        s += v; q += v * v;
    }
    ssum[threadIdx.x] = s; ssq[threadIdx.x] = q;
    __syncthreads();
    for (int o = 128; o > 0; o >>= 1) {
        if (threadIdx.x < o) { ssum[threadIdx.x] += ssum[threadIdx.x + o];
                               ssq[threadIdx.x]  += ssq[threadIdx.x + o]; }
        __syncthreads();
    }
    if (threadIdx.x == 0) {
        float mu  = ssum[0] / rows;
        float var = ssq[0] / rows - mu * mu;
        float sc  = gamma[c] * rsqrtf(var + 1e-5f);
        scale[c] = sc;
        shift[c] = beta[c] - mu * sc;
    }
}

// ---------------- host side ----------------
static inline unsigned blks(long n, int b) { return (unsigned)((n + b - 1) / b); }

extern "C" void kernel_launch(void* const* d_in, const int* in_sizes, int n_in,
                              void* d_out, int out_size, void* d_ws, size_t ws_size,
                              hipStream_t stream)
{
    (void)in_sizes; (void)n_in; (void)out_size; (void)ws_size;

    const float* x1 = (const float*)d_in[0];
    const float* x2 = (const float*)d_in[1];
    const int* ei1 = (const int*)d_in[2];
    const int* ei2 = (const int*)d_in[3];
    const int* b1  = (const int*)d_in[4];
    const int* b2  = (const int*)d_in[5];
    const float* notes1 = (const float*)d_in[6];
    const float* notes2 = (const float*)d_in[7];
    const float* Wc1 = (const float*)d_in[8];  const float* bc1 = (const float*)d_in[9];
    const float* Wc2 = (const float*)d_in[10]; const float* bc2 = (const float*)d_in[11];
    const float* Wc3 = (const float*)d_in[12]; const float* bc3 = (const float*)d_in[13];
    const float* Wc4 = (const float*)d_in[14]; const float* bc4 = (const float*)d_in[15];
    const float* Wn1 = (const float*)d_in[16]; const float* bn1 = (const float*)d_in[17];
    const float* Wn2 = (const float*)d_in[18]; const float* bn2 = (const float*)d_in[19];
    const float* Wf1 = (const float*)d_in[20]; const float* bf1 = (const float*)d_in[21];
    const float* g1  = (const float*)d_in[22]; const float* be1 = (const float*)d_in[23];
    const float* Wf2 = (const float*)d_in[24]; const float* bf2 = (const float*)d_in[25];
    const float* g2  = (const float*)d_in[26]; const float* be2 = (const float*)d_in[27];
    const float* Wf3 = (const float*)d_in[28]; const float* bf3 = (const float*)d_in[29];

    // ---- workspace carving ----
    char* p = (char*)d_ws;
    auto carve = [&](size_t bytes) -> void* {
        void* r = (void*)p; p += (bytes + 255) & ~(size_t)255; return r;
    };
    float*  hA     = (float*) carve((size_t)NN * HID * 4);
    float*  hB     = (float*) carve((size_t)NN * HID * 4);
    __bf16* actbf  = (__bf16*)carve((size_t)NN * HID * 2);
    float*  dinv   = (float*) carve((size_t)NN * 4);
    float*  concat = (float*) carve((size_t)GG * 512 * 4);
    float*  smallf = (float*) carve((size_t)GG * 256 * 4);
    __bf16* smallbf= (__bf16*)carve((size_t)GG * 512 * 2);
    float*  cnt    = (float*) carve((size_t)GG * 4);
    float*  bnsc   = (float*) carve(256 * 4);
    float*  bnsh   = (float*) carve(256 * 4);
    __bf16* wts    = (__bf16*)carve((size_t)442368 * 2);

    __bf16* Wc1t = wts +      0;  // [256,128]
    __bf16* Wc2t = wts +  32768;  // [256,256]
    __bf16* Wc3t = wts +  98304;  // [256,256]
    __bf16* Wc4t = wts + 163840;  // [128,256]
    __bf16* Wn1t = wts + 196608;  // [256,128]
    __bf16* Wn2t = wts + 229376;  // [128,256]
    __bf16* Wf1t = wts + 262144;  // [256,512]
    __bf16* Wf2t = wts + 393216;  // [128,256]
    __bf16* Wf3t = wts + 425984;  // [128,128]

    // ---- helpers ----
    auto cvt = [&](const float* x, __bf16* y, long n) {
        k_cvt<<<blks(n / 4, 256), 256, 0, stream>>>(x, y, n / 4);
    };
    auto gemm = [&](const __bf16* A, const __bf16* Bt, const float* bias, float* C,
                    int M, int K, int Nc, int ldc, int relu) {
        dim3 g(blks(M, BM), Nc / BN);
        k_gemm_bf16<<<g, 256, 0, stream>>>(A, Bt, bias, C, M, K, ldc, relu);
    };
    auto wt = [&](const float* W, __bf16* Wt, int K, int Nc) {
        k_wt<<<blks((long)K * Nc, 256), 256, 0, stream>>>(W, Wt, K, Nc);
    };

    // ---- convert weights (transposed, bf16) ----
    wt(Wc1, Wc1t, IND, HID);  wt(Wc2, Wc2t, HID, HID);
    wt(Wc3, Wc3t, HID, HID);  wt(Wc4, Wc4t, HID, H2);
    wt(Wn1, Wn1t, NL,  HID);  wt(Wn2, Wn2t, HID, H2);
    wt(Wf1, Wf1t, 4*H2, HID); wt(Wf2, Wf2t, HID, H2);
    wt(Wf3, Wf3t, H2,  NL);

    hipMemsetAsync(concat, 0, (size_t)GG * 512 * 4, stream);

    // ---- GCN tower ----
    auto tower = [&](const float* x, const int* ei, const int* batch, int off) {
        // symmetric normalization
        k_fill<<<blks(NN, 256), 256, 0, stream>>>(dinv, 1.0f, NN);      // self loop
        k_deg<<<blks(EE, 256), 256, 0, stream>>>(ei + EE, dinv, EE);
        k_rsqrt<<<blks(NN, 256), 256, 0, stream>>>(dinv, NN);

        const __bf16* Wt4[4]   = {Wc1t, Wc2t, Wc3t, Wc4t};
        const float*  bias4[4] = {bc1, bc2, bc3, bc4};
        int Kdim[4] = {IND, HID, HID, HID};
        int Ndim[4] = {HID, HID, HID, H2};

        for (int L = 0; L < 4; ++L) {
            int K = Kdim[L], C = Ndim[L];
            int cl2 = (C == 256) ? 8 : 7;
            const float* src = (L == 0) ? x : hA;
            cvt(src, actbf, (long)NN * K);
            gemm(actbf, Wt4[L], nullptr, hA, NN, K, C, C, 0);
            hipMemsetAsync(hB, 0, (size_t)NN * C * 4, stream);
            long sct = (long)EE * (C >> 2);
            k_scatter<<<blks(sct, 256), 256, 0, stream>>>(
                hA, ei, ei + EE, dinv, hB, EE, cl2 - 2, (C >> 2) - 1, C);
            long n4 = (long)NN * C / 4;
            k_selfbias<<<blks(n4, 256), 256, 0, stream>>>(
                hA, hB, dinv, bias4[L], n4, cl2 - 2, (C >> 2) - 1);
        }
        // mean pool into concat slice [*, off:off+128]
        hipMemsetAsync(cnt, 0, (size_t)GG * 4, stream);
        long pt = (long)NN * 32;
        k_pool_sum<<<blks(pt, 256), 256, 0, stream>>>(hA, batch, concat + off, cnt, NN);
        k_pool_div<<<blks((long)GG * 128, 256), 256, 0, stream>>>(concat + off, cnt);
    };

    tower(x1, ei1, b1, 0);
    tower(x2, ei2, b2, 128);

    // ---- notes MLPs -> concat slices ----
    auto notes_mlp = [&](const float* nt, int off) {
        cvt(nt, smallbf, (long)GG * NL);
        gemm(smallbf, Wn1t, bn1, smallf, GG, NL, HID, HID, 1);
        cvt(smallf, smallbf, (long)GG * HID);
        gemm(smallbf, Wn2t, bn2, concat + off, GG, HID, H2, 512, 1);
    };
    notes_mlp(notes1, 256);
    notes_mlp(notes2, 384);

    // ---- fusion head ----
    cvt(concat, smallbf, (long)GG * 512);
    gemm(smallbf, Wf1t, bf1, smallf, GG, 512, HID, HID, 1);             // relu(c@Wf1+bf1)
    k_bn_stats<<<HID, 256, 0, stream>>>(smallf, g1, be1, bnsc, bnsh, GG, HID);
    k_cvt_affine<<<blks((long)GG * HID / 4, 256), 256, 0, stream>>>(
        smallf, bnsc, bnsh, smallbf, (long)GG * HID / 4, HID - 1);      // BN fused into cvt
    gemm(smallbf, Wf2t, bf2, smallf, GG, HID, H2, H2, 1);               // relu(h@Wf2+bf2)
    k_bn_stats<<<H2, 256, 0, stream>>>(smallf, g2, be2, bnsc, bnsh, GG, H2);
    k_cvt_affine<<<blks((long)GG * H2 / 4, 256), 256, 0, stream>>>(
        smallf, bnsc, bnsh, smallbf, (long)GG * H2 / 4, H2 - 1);
    gemm(smallbf, Wf3t, bf3, (float*)d_out, GG, H2, NL, NL, 0);         // final, no relu
}
